// Decoder_74938589381001
// MI455X (gfx1250) — matrix operand order repaired and checked by reading, not compiled, not run on hardware
//
#include <hip/hip_runtime.h>
#include <hip/hip_bf16.h>
#include <stdint.h>

// ---------------------------------------------------------------------------
// MI455X (gfx1250) implementation, round 2.
// conv1d == implicit GEMM (M=Cout, N=B*T, K=Cin per tap) on v_wmma_f32_16x16x32_bf16.
// - Weights repacked at launch into padded [Kw][CoutP][CinP] bf16 (CoutP %128==0,
//   CinP %32==0, zero filled) -> A staging is pure b128 traffic, no bounds checks.
// - B staging: one column per thread (coalesced f32 loads), shift-based b/t math
//   (all Tout are powers of two), bf16 pack, 2x ds_store_b128.
// - Double-buffered LDS, one barrier per K-chunk; WMMAs hide staging latency.
// - Fused: pre-ReLU, post-ReLU, bias, residual add, nearest-x2 upsample.
// ---------------------------------------------------------------------------

typedef __attribute__((ext_vector_type(16))) __bf16 v16bf;
typedef __attribute__((ext_vector_type(8)))  float  v8f;

#define TILE_M 128
#define TILE_N 128
#define KC 32
#define LDSS 40      // LDS row stride in halves (KC + 8 pad, spreads banks)

__device__ __forceinline__ unsigned short f32_to_bf16_rne(float f) {
  unsigned int u = __float_as_uint(f);
  u += 0x7FFFu + ((u >> 16) & 1u);
  return (unsigned short)(u >> 16);
}

union FragU { uint4 q[2]; v16bf v; };

// Repack weight [Cout,Cin,Kw] f32 -> [Kw][CoutP][CinP] bf16, zero padded.
__global__ void pack_w_bf16(const float* __restrict__ src,
                            unsigned short* __restrict__ dst,
                            int Cout, int Cin, int Kw, int CoutP, int CinP) {
  long i = (long)blockIdx.x * 256 + threadIdx.x;
  long tot = (long)Kw * CoutP * CinP;
  if (i >= tot) return;
  int ci = (int)(i % CinP);
  long r = i / CinP;
  int co = (int)(r % CoutP);
  int k  = (int)(r / CoutP);
  unsigned short v = 0;
  if (co < Cout && ci < Cin)
    v = f32_to_bf16_rne(src[((long)co * Cin + ci) * Kw + k]);
  dst[i] = v;
}

// Average nodes (b=0,n=0) and (b=0,n=1) of the GCN output.
__global__ void gcn_mix(float* __restrict__ h, int n) {
  int i = blockIdx.x * 256 + threadIdx.x;
  if (i < n) {
    float m = 0.5f * (h[i] + h[n + i]);
    h[i] = m;
    h[n + i] = m;
  }
}

// Fused conv1d as implicit GEMM. Host guarantees: Tout == 1<<tShift,
// Bn*Tout is a multiple of TILE_N, wpack padded to [Kw][CoutP][CinP].
__global__ __launch_bounds__(256)
void conv1d_wmma_bf16(const float* __restrict__ x,
                      const unsigned short* __restrict__ wpack,
                      const float* __restrict__ bias,
                      const float* __restrict__ residual,
                      float* __restrict__ y,
                      int Cin, int Cout, int CoutP, int CinP,
                      int Tin, int Tout, int tShift,
                      int Kw, int dil, int ups, int preRelu, int postRelu,
                      long xStrideB, long yStrideB, long resStrideB) {
  __shared__ unsigned short lsA[2][TILE_M * LDSS];
  __shared__ unsigned short lsB[2][TILE_N * LDSS];

  const int tid  = threadIdx.x;
  const int lane = tid & 31;
  const int wv   = tid >> 5;          // wave id 0..7
  const int g    = lane >> 4;         // lane half-group
  const int ln   = lane & 15;

  const int wm0 = (wv >> 2) * 64;     // wave M offset (2 waves in M)
  const int wn0 = (wv & 3) * 32;      // wave N offset (4 waves in N)

  const int m0  = blockIdx.y * TILE_M;
  const int n0  = blockIdx.x * TILE_N;
  const int pad = dil * (Kw - 1) / 2;

  // ---- per-thread staging geometry (fixed for the whole kernel) ----
  const int sr = tid & 127;           // row within tile (mr for A, nr for B)
  const int kq = tid >> 7;            // 0/1 : which 16-half segment of KC
  // B column -> (b, t) once
  const int nB = n0 + sr;
  const int bB = nB >> tShift;
  const int tB = nB & (Tout - 1);
  const float* xcol = x + (long)bB * xStrideB;

  auto stage = [&](int bi, int k, int c0) {
    // A: weights, contiguous bf16, no bounds checks (padded layout)
    {
      const unsigned short* src =
          wpack + ((long)k * CoutP + (m0 + sr)) * CinP + c0 + kq * 16;
      uint4 q0 = *(const uint4*)(src);
      uint4 q1 = *(const uint4*)(src + 8);
      unsigned short* d = &lsA[bi][sr * LDSS + kq * 16];
      *(uint4*)(d)     = q0;
      *(uint4*)(d + 8) = q1;
      if (c0 + KC < CinP)   // prefetch next weight chunk (global_prefetch_b8)
        __builtin_prefetch(src + KC, 0, 3);
    }
    // B: activations, one column per thread, 16 consecutive ci
    {
      int ti = tB - pad + k * dil;
      bool valid;
      int tsrc;
      if (ups) { valid = (ti >= 0) && (ti < 2 * Tin); tsrc = valid ? (ti >> 1) : 0; }
      else     { valid = (ti >= 0) && (ti < Tin);     tsrc = valid ? ti        : 0; }
      const int ci0 = c0 + kq * 16;
      const float* xp = xcol + (long)ci0 * Tin + tsrc;
      unsigned int u[8];
#pragma unroll
      for (int j = 0; j < 8; ++j) {
        int ciA = ci0 + 2 * j, ciB = ciA + 1;
        float f0 = (valid && ciA < Cin) ? xp[(long)(2 * j) * Tin] : 0.f;
        float f1 = (valid && ciB < Cin) ? xp[(long)(2 * j + 1) * Tin] : 0.f;
        if (preRelu) { f0 = fmaxf(f0, 0.f); f1 = fmaxf(f1, 0.f); }
        u[j] = (unsigned)f32_to_bf16_rne(f0) |
               ((unsigned)f32_to_bf16_rne(f1) << 16);
      }
      uint4 q0 = {u[0], u[1], u[2], u[3]};
      uint4 q1 = {u[4], u[5], u[6], u[7]};
      unsigned short* d = &lsB[bi][sr * LDSS + kq * 16];
      *(uint4*)(d)     = q0;
      *(uint4*)(d + 8) = q1;
    }
  };

  const v8f vzero = {0.f, 0.f, 0.f, 0.f, 0.f, 0.f, 0.f, 0.f};
  v8f acc[4][2];
#pragma unroll
  for (int mt = 0; mt < 4; ++mt)
#pragma unroll
    for (int nf = 0; nf < 2; ++nf) acc[mt][nf] = vzero;

  const int ckPer = CinP >> 5;
  const int nch = Kw * ckPer;

  // next-chunk (k, c0) tracker for the staging pipeline
  int sk = 0, sc0 = 0;
  stage(0, sk, sc0);
  sc0 += KC; if (sc0 == CinP) { sc0 = 0; ++sk; }
  __syncthreads();

  int buf = 0;
  for (int ch = 0; ch < nch; ++ch) {
    if (ch + 1 < nch) {
      stage(buf ^ 1, sk, sc0);
      sc0 += KC; if (sc0 == CinP) { sc0 = 0; ++sk; }
    }
    // ---- compute from lsA/lsB[buf]: 8 WMMA per wave ----
    FragU af[4], bfr[2];
#pragma unroll
    for (int mt = 0; mt < 4; ++mt) {
      int mrow = wm0 + mt * 16 + ln;
      const unsigned short* p = &lsA[buf][mrow * LDSS];
      af[mt].q[0] = *(const uint4*)(p + 8 * g);        // j<8 : K = 8g+j
      af[mt].q[1] = *(const uint4*)(p + 16 + 8 * g);   // j>=8: K = 16+8g+(j-8)
    }
#pragma unroll
    for (int nf = 0; nf < 2; ++nf) {
      int nrow = wn0 + nf * 16 + ln;
      const unsigned short* p = &lsB[buf][nrow * LDSS];
      bfr[nf].q[0] = *(const uint4*)(p + 8 * g);       // j<8 : K = 8g+j
      bfr[nf].q[1] = *(const uint4*)(p + 16 + 8 * g);  // j>=8: K = 16+8g+(j-8)
    }
#pragma unroll
    for (int nf = 0; nf < 2; ++nf)
#pragma unroll
      for (int mt = 0; mt < 4; ++mt)
        acc[mt][nf] = __builtin_amdgcn_wmma_f32_16x16x32_bf16(
            false, af[mt].v, false, bfr[nf].v, (short)0, acc[mt][nf],
            false, false);
    __syncthreads();
    buf ^= 1;
  }

  // ---- epilogue: bias + residual + relu + store (coalesced over ln->t) ----
#pragma unroll
  for (int mt = 0; mt < 4; ++mt) {
#pragma unroll
    for (int i = 0; i < 8; ++i) {
      int co = m0 + wm0 + mt * 16 + i + 8 * g;         // D: M = i + 8*g
      if (co < Cout) {
        float bv = bias[co];
#pragma unroll
        for (int nf = 0; nf < 2; ++nf) {
          int n = n0 + wn0 + nf * 16 + ln;             // D: N = lane&15
          int b = n >> tShift, t = n & (Tout - 1);
          float v = acc[mt][nf][i] + bv;
          if (residual) v += residual[(long)b * resStrideB + (long)co * Tout + t];
          if (postRelu) v = fmaxf(v, 0.f);
          y[(long)b * yStrideB + (long)co * Tout + t] = v;
        }
      }
    }
  }
}

// ---------------------------------------------------------------------------
// Host orchestration
// ---------------------------------------------------------------------------

static inline int roundUp(int v, int m) { return (v + m - 1) / m * m; }

extern "C" void kernel_launch(void* const* d_in, const int* in_sizes, int n_in,
                              void* d_out, int out_size, void* d_ws, size_t ws_size,
                              hipStream_t stream) {
  (void)in_sizes; (void)n_in; (void)out_size; (void)ws_size;
  const int BATCH = 32, W = 512, T0 = 64;
  const int GOUT = 156, UF = 156, LF = 107;

  // Input leaf indices assume insertion-order flattening of setup_inputs():
  // 0 ux, 1 lx, [2..49] ubody_params, [50..97] lbody_params,
  // 98 gcn_w, 99 gcn_b, 100 uconv_w, 101 uconv_b, 102 lconv_w, 103 lconv_b.
  // Branch leaves: in_w,in_b, 3x{ 3x{w1,b1,w2,b2}, up_w,up_b }, out1..out2.

  unsigned short* warena = (unsigned short*)d_ws;
  size_t woff = 0;
  auto packW = [&](int idx, int Cout, int Cin, int Kw) -> const unsigned short* {
    int CoutP = roundUp(Cout, TILE_M), CinP = roundUp(Cin, KC);
    long tot = (long)Kw * CoutP * CinP;
    unsigned short* dst = warena + woff;
    pack_w_bf16<<<dim3((unsigned)((tot + 255) / 256)), dim3(256), 0, stream>>>(
        (const float*)d_in[idx], dst, Cout, Cin, Kw, CoutP, CinP);
    woff += (size_t)((tot + 7) & ~7l);
    return dst;
  };
  auto F = [&](int idx) -> const float* { return (const float*)d_in[idx]; };

  struct ResW { const unsigned short* w1; const float* b1;
                const unsigned short* w2; const float* b2; };
  struct BlkW { ResW res[3]; const unsigned short* up_w; const float* up_b; };
  struct BranchW { const unsigned short* in_w; const float* in_b; BlkW blk[3];
                   const unsigned short* o1w; const float* o1b;
                   const unsigned short* o2w; const float* o2b; };

  auto loadBranch = [&](int base) -> BranchW {
    BranchW br;
    br.in_w = packW(base + 0, W, W, 3); br.in_b = F(base + 1);
    for (int b = 0; b < 3; ++b) {
      int bb = base + 2 + b * 14;
      for (int r = 0; r < 3; ++r) {
        int rb = bb + r * 4;
        br.blk[b].res[r].w1 = packW(rb + 0, W, W, 3); br.blk[b].res[r].b1 = F(rb + 1);
        br.blk[b].res[r].w2 = packW(rb + 2, W, W, 1); br.blk[b].res[r].b2 = F(rb + 3);
      }
      br.blk[b].up_w = packW(bb + 12, W, W, 3); br.blk[b].up_b = F(bb + 13);
    }
    br.o1w = packW(base + 44, W, W, 3); br.o1b = F(base + 45);
    br.o2w = packW(base + 46, W, W, 3); br.o2b = F(base + 47);
    return br;
  };

  BranchW ub = loadBranch(2);
  BranchW lb = loadBranch(50);
  const unsigned short* gcn_w = packW(98,  GOUT, W, 1);    const float* gcn_b = F(99);
  const unsigned short* uc_w  = packW(100, UF, GOUT, 3);   const float* uc_b  = F(101);
  const unsigned short* lc_w  = packW(102, LF, GOUT, 3);   const float* lc_b  = F(103);

  // Activation workspace: 4 rotating f32 buffers of [32,512,512].
  size_t actoff = ((woff * 2 + 255) & ~(size_t)255);
  const size_t ACT = (size_t)BATCH * W * 512;   // 8,388,608 floats each
  float* bufs[4];
  for (int i = 0; i < 4; ++i)
    bufs[i] = (float*)((char*)d_ws + actoff + (size_t)i * ACT * sizeof(float));

  auto conv = [&](const float* xx, const unsigned short* ww, const float* bb,
                  const float* rr, float* yy,
                  int Cin, int Cout, int Tin, int Tout, int Kw, int dil,
                  int ups, int pre, int post, long xsb, long ysb, long rsb) {
    int CoutP = roundUp(Cout, TILE_M), CinP = roundUp(Cin, KC);
    int tShift = __builtin_ctz(Tout);             // all Tout are powers of two
    dim3 grid((unsigned)((BATCH * Tout) / TILE_N),
              (unsigned)(CoutP / TILE_M));
    conv1d_wmma_bf16<<<grid, dim3(256), 0, stream>>>(
        xx, ww, bb, rr, yy, Cin, Cout, CoutP, CinP, Tin, Tout, tShift,
        Kw, dil, ups, pre, post, xsb, ysb, rsb);
  };

  static const int DILS[3] = {9, 3, 1};

  auto runBranch = [&](const float* x_in, const BranchW& br,
                       float* P0, float* P1, float* P2) -> float* {
    float* P[3] = {P0, P1, P2};
    int T = T0;
    conv(x_in, br.in_w, br.in_b, nullptr, P[0], W, W, T, T, 3, 1,
         0, 0, 1, (long)W * T, (long)W * T, 0);
    int xi = 0;
    for (int b = 0; b < 3; ++b) {
      for (int r = 0; r < 3; ++r) {
        int hi = (xi + 1) % 3, yi = (xi + 2) % 3;
        conv(P[xi], br.blk[b].res[r].w1, br.blk[b].res[r].b1, nullptr, P[hi],
             W, W, T, T, 3, DILS[r], 0, 1, 1, (long)W * T, (long)W * T, 0);
        conv(P[hi], br.blk[b].res[r].w2, br.blk[b].res[r].b2, P[xi], P[yi],
             W, W, T, T, 1, 1, 0, 0, 0, (long)W * T, (long)W * T, (long)W * T);
        xi = yi;
      }
      int ui = (xi + 1) % 3;
      conv(P[xi], br.blk[b].up_w, br.blk[b].up_b, nullptr, P[ui],
           W, W, T, 2 * T, 3, 1, 1, 0, 0, (long)W * T, (long)W * 2 * T, 0);
      T *= 2; xi = ui;
    }
    int o1 = (xi + 1) % 3;
    conv(P[xi], br.o1w, br.o1b, nullptr, P[o1], W, W, T, T, 3, 1,
         0, 0, 1, (long)W * T, (long)W * T, 0);
    int o2 = (o1 + 1) % 3;
    conv(P[o1], br.o2w, br.o2b, nullptr, P[o2], W, W, T, T, 3, 1,
         0, 0, 0, (long)W * T, (long)W * T, 0);
    return P[o2];   // deterministically P[2] of this pool
  };

  float* u_out = runBranch((const float*)d_in[0], ub, bufs[0], bufs[1], bufs[2]);
  float* poolL[3]; int pi = 0;
  for (int i = 0; i < 4; ++i) if (bufs[i] != u_out && pi < 3) poolL[pi++] = bufs[i];
  float* l_out = runBranch((const float*)d_in[1], lb, poolL[0], poolL[1], poolL[2]);
  float* hbuf = nullptr;
  for (int i = 0; i < 4; ++i)
    if (bufs[i] != u_out && bufs[i] != l_out) { hbuf = bufs[i]; break; }

  const int TF = 512;
  // GCN linear per node: h[:,node,:,:] = gcn_w @ branch + gcn_b  (k=1 GEMM)
  conv(u_out, gcn_w, gcn_b, nullptr, hbuf, W, GOUT, TF, TF, 1, 1,
       0, 0, 0, (long)W * TF, (long)2 * GOUT * TF, 0);
  conv(l_out, gcn_w, gcn_b, nullptr, hbuf + (size_t)GOUT * TF, W, GOUT,
       TF, TF, 1, 1, 0, 0, 0, (long)W * TF, (long)2 * GOUT * TF, 0);

  // Mix the two batch-0 nodes (edge_idx = [[0,1],[1,0]]).
  {
    int n = GOUT * TF;
    gcn_mix<<<dim3((n + 255) / 256), dim3(256), 0, stream>>>(hbuf, n);
  }

  // Final convs straight into d_out (tuple: ux_out then lx_out).
  float* out_u = (float*)d_out;
  float* out_l = out_u + (size_t)BATCH * UF * TF;
  conv(hbuf, uc_w, uc_b, nullptr, out_u, GOUT, UF, TF, TF, 3, 1,
       0, 0, 0, (long)2 * GOUT * TF, (long)UF * TF, 0);
  conv(hbuf + (size_t)GOUT * TF, lc_w, lc_b, nullptr, out_l, GOUT, LF,
       TF, TF, 3, 1, 0, 0, 0, (long)2 * GOUT * TF, (long)LF * TF, 0);
}
